// GraphSAGE_80023830659403
// MI455X (gfx1250) — matrix operand order repaired and checked by reading, not compiled, not run on hardware
//
#include <hip/hip_runtime.h>
#include <stdint.h>

// ---------------------------------------------------------------------------
// GraphSAGE (3x SAGEConv with 3x3 convs, relu, maxpool2) for gfx1250 / MI455X.
// Convs: implicit GEMM on v_wmma_f32_16x16x32_bf16 (f32 accumulate), K-loop
// fully unrolled so im2col offsets fold to immediate ds_load offsets.
// LDS staging via Tensor Data Mover (tensor_load_to_lds + s_wait_tensorcnt).
// Aggregation: atomic scatter-add (bandwidth path, ~1.3GB @ 23.3TB/s).
// ---------------------------------------------------------------------------

#define NN 2000
#define NE 32000

typedef __attribute__((ext_vector_type(16))) __bf16 v16bf;
typedef __attribute__((ext_vector_type(8)))  float  v8f;
typedef uint32_t tdm_u32x4 __attribute__((ext_vector_type(4)));
typedef int      tdm_i32x4 __attribute__((ext_vector_type(4)));
typedef int      tdm_i32x8 __attribute__((ext_vector_type(8)));

#if defined(__has_builtin)
#if __has_builtin(__builtin_amdgcn_tensor_load_to_lds) && \
    __has_builtin(__builtin_amdgcn_s_wait_tensorcnt)
#define USE_TDM 1
#endif
#endif
#ifndef USE_TDM
#define USE_TDM 0
#endif

// --------------------------- utility kernels -------------------------------

__global__ void zero_kernel(float* __restrict__ p, size_t n) {
  size_t i = (size_t)blockIdx.x * blockDim.x + threadIdx.x;
  size_t stride = (size_t)gridDim.x * blockDim.x;
  for (; i < n; i += stride) p[i] = 0.0f;
}

// zero F floats per node at stride ns
__global__ void zero_strided_kernel(float* __restrict__ p, size_t ns, int F) {
  int n = blockIdx.y;
  int f = blockIdx.x * blockDim.x + threadIdx.x;
  if (f < F) p[(size_t)n * ns + f] = 0.0f;
}

__global__ void degree_kernel(const int* __restrict__ dst, float* __restrict__ deg) {
  int e = blockIdx.x * blockDim.x + threadIdx.x;
  if (e < NE) atomicAdd(&deg[dst[e]], 1.0f);
}

// copy F floats per node: src[n*ss + f] -> dst[n*ds + f]
__global__ void copy_strided_kernel(const float* __restrict__ src, size_t ss,
                                    float* __restrict__ dst, size_t ds, int F) {
  int n = blockIdx.y;
  int f = blockIdx.x * blockDim.x + threadIdx.x;
  if (f < F) dst[(size_t)n * ds + f] = src[(size_t)n * ss + f];
}

// for each edge e: agg[dst[e]] += x[src[e]]  (F floats)
__global__ void edge_agg_kernel(const float* __restrict__ x, size_t xs, int F,
                                float* __restrict__ agg, size_t as_,
                                const int* __restrict__ src,
                                const int* __restrict__ dst) {
  int e = blockIdx.x;
  int s = src[e], d = dst[e];
  const float* xp = x + (size_t)s * xs;
  float* ap = agg + (size_t)d * as_;
  int f = blockIdx.y * blockDim.x + threadIdx.x;
  if (f < F) atomicAdd(&ap[f], xp[f]);
}

__global__ void scale_agg_kernel(float* __restrict__ agg, size_t ns, int F,
                                 const float* __restrict__ deg) {
  int n = blockIdx.y;
  float inv = 1.0f / fmaxf(deg[n], 1.0f);
  int f = blockIdx.x * blockDim.x + threadIdx.x;
  if (f < F) agg[(size_t)n * ns + f] *= inv;
}

// Pack [w_self ; w_neigh] ([Cout,C,3,3] f32 each) into bf16 GEMM-A [Cout x Kpad],
// row-major, K = ci_cat*9 + ky*3 + kx with ci_cat in [0,2C); zero-padded to Kpad.
__global__ void pack_w_kernel(const float* __restrict__ wself,
                              const float* __restrict__ wneigh,
                              __bf16* __restrict__ out, int Cout, int C, int Kpad) {
  int co = blockIdx.y;
  int k  = blockIdx.x * blockDim.x + threadIdx.x;
  if (k >= Kpad) return;
  float v = 0.0f;
  int K2 = 2 * C * 9;
  if (k < K2) {
    int ci = k / 9, r = k % 9;
    v = (ci < C) ? wself[((size_t)co * C + ci) * 9 + r]
                 : wneigh[((size_t)co * C + (ci - C)) * 9 + r];
  }
  out[(size_t)co * Kpad + k] = (__bf16)v;
}

// 2x2/2 maxpool: in [NN][C][Ho][Wo] -> out[n*ons + c*ocs + py*PW + px]
__global__ void pool_kernel(const float* __restrict__ in, float* __restrict__ out,
                            int C, int Ho, int Wo, size_t ons, int ocs) {
  int PH = Ho >> 1, PW = Wo >> 1;
  size_t total = (size_t)NN * C * PH * PW;
  size_t i = (size_t)blockIdx.x * blockDim.x + threadIdx.x;
  size_t stride = (size_t)gridDim.x * blockDim.x;
  for (; i < total; i += stride) {
    int px = (int)(i % PW);
    int py = (int)((i / PW) % PH);
    int c  = (int)((i / ((size_t)PW * PH)) % C);
    int n  = (int)(i / ((size_t)PW * PH * C));
    const float* ip = in + ((size_t)n * C + c) * Ho * Wo;
    int y = 2 * py, x = 2 * px;
    float a = ip[y * Wo + x],        b = ip[y * Wo + x + 1];
    float cc = ip[(y + 1) * Wo + x], d = ip[(y + 1) * Wo + x + 1];
    out[(size_t)n * ons + (size_t)c * ocs + py * PW + px] =
        fmaxf(fmaxf(a, b), fmaxf(cc, d));
  }
}

// -------------------- TDM staging: global f32 row -> LDS -------------------
// One wave issues a 1-row tensor_load_to_lds (nElem f32), waits TENSORcnt==0,
// then the workgroup barrier publishes LDS contents to all waves.
__device__ __forceinline__ void tdm_load_row(const void* gptr, uint32_t lds_byte,
                                             uint32_t nElem) {
#if USE_TDM
  uint64_t ga = (uint64_t)(uintptr_t)gptr;
  tdm_u32x4 g0;
  g0[0] = 1u;                                           // count=1 (valid D#)
  g0[1] = lds_byte;                                     // lds_addr
  g0[2] = (uint32_t)ga;                                 // global_addr[31:0]
  g0[3] = (uint32_t)((ga >> 32) & 0x01ffffffu) | (2u << 30);  // addr[56:32]|type=2
  tdm_i32x8 g1;
  g1[0] = (int)(2u << 16);                              // data_size = 4 bytes
  g1[1] = (int)((nElem & 0xffffu) << 16);               // tensor_dim0 lo
  g1[2] = (int)(((nElem >> 16) & 0xffffu) | (1u << 16)); // tensor_dim0 hi | dim1=1
  g1[3] = (int)((nElem & 0xffffu) << 16);               // tile_dim0 = nElem
  g1[4] = 1;                                            // tile_dim1=1, tile_dim2=0
  g1[5] = (int)nElem;                                   // tensor_dim0_stride lo
  g1[6] = (int)((nElem & 0xffffu) << 16);               // stride0 hi=0 | stride1 lo
  g1[7] = (int)(nElem >> 16);                           // stride1 hi
  tdm_i32x4 g2 = {0, 0, 0, 0};
  tdm_i32x4 g3 = {0, 0, 0, 0};
#if __clang_major__ >= 23
  tdm_i32x8 g4 = {0, 0, 0, 0, 0, 0, 0, 0};
  __builtin_amdgcn_tensor_load_to_lds(g0, g1, g2, g3, g4, 0);
#else
  __builtin_amdgcn_tensor_load_to_lds(g0, g1, g2, g3, 0);
#endif
  __builtin_amdgcn_s_wait_tensorcnt(0);
#else
  (void)gptr; (void)lds_byte; (void)nElem;
#endif
}

// --------------------------- WMMA conv kernel ------------------------------
// One workgroup (8 waves) per node. xcat[n] ([C2][H][W] f32) staged in LDS via
// TDM. Each wave computes 16(cout) x 16(spatial) tiles with
// v_wmma_f32_16x16x32_bf16; the K loop is fully unrolled so every im2col
// offset folds to an immediate ds_load offset.
template <int C2, int H, int W, int Cout, int Kpad, int Ho, int Wo>
__global__ __launch_bounds__(256) void conv_wmma_kernel(
    const float* __restrict__ xcat, const __bf16* __restrict__ Aw,
    float* __restrict__ outBuf) {
  constexpr int IN    = C2 * H * W;
  constexpr int HoWo  = Ho * Wo;
  constexpr int Mt    = Cout / 16;
  constexpr int Nt    = (HoWo + 15) / 16;
  constexpr int KREAL = C2 * 9;

  __shared__ float ldsIn[IN];

  const int n = blockIdx.x;
  const float* xin = xcat + (size_t)n * IN;

#if USE_TDM
  if (threadIdx.x < 32) {   // wave 0 issues the DMA; EXEC ignored by TDM
    tdm_load_row(xin, (uint32_t)(uintptr_t)&ldsIn[0], (uint32_t)IN);
  }
#else
  for (int i = threadIdx.x; i < IN; i += blockDim.x) ldsIn[i] = xin[i];
#endif
  __syncthreads();

  const int wave   = threadIdx.x >> 5;
  const int lane   = threadIdx.x & 31;
  const int lhalf  = (lane < 16) ? 0 : 1;
  const int ln     = lane & 15;
  const int nwaves = blockDim.x >> 5;

  for (int t = wave; t < Mt * Nt; t += nwaves) {   // wave-uniform loop
    const int m0 = (t / Nt) * 16;
    const int p  = (t % Nt) * 16 + ln;             // output spatial position
    const int pc = (p < HoWo) ? p : (HoWo - 1);    // clamp (value-select only)
    const int base = (pc / Wo) * W + (pc % Wo);    // im2col base in ldsIn

    v8f acc = {};
    const __bf16* arow = Aw + (size_t)(m0 + ln) * Kpad;

#pragma unroll
    for (int k0 = 0; k0 < Kpad; k0 += 32) {
      // A fragment: this code's contiguous k order (same as B). Lane (m = m0+ln):
      //   elems 0..7  -> K = k0 + kh + j  (kh = 0/16 by half-wave)
      //   elems 8..15 -> K = k0 + kh + 8 + j
      v16bf a;
      const __bf16* ap = arow + k0 + lhalf * 16;
#pragma unroll
      for (int j = 0; j < 8; ++j) {
        a[j]     = ap[j];        // contiguous 16B group -> b128 load
        a[j + 8] = ap[8 + j];    // contiguous 16B group -> b128 load
      }
      // B fragment: 32x16 layout. Column = ln; lanes<16 hold K=k0+j,
      // lanes>=16 hold K=k0+16+j. Split by half-wave so each side's
      // (K/9, K%9) folds to an immediate LDS offset.
      v16bf b;
      if (lane < 16) {
#pragma unroll
        for (int j = 0; j < 16; ++j) {
          const int K  = k0 + j;                    // unroll-constant
          const int o  = (K / 9) * H * W + ((K % 9) / 3) * W + (K % 3);
          b[j] = (__bf16)ldsIn[base + o];
        }
      } else {
#pragma unroll
        for (int j = 0; j < 16; ++j) {
          const int K = k0 + 16 + j;                // unroll-constant
          float v = 0.0f;
          if (K < KREAL) {                          // compile-time per j
            const int o = (K / 9) * H * W + ((K % 9) / 3) * W + (K % 3);
            v = ldsIn[base + o];
          }
          b[j] = (__bf16)v;
        }
      }
      acc = __builtin_amdgcn_wmma_f32_16x16x32_bf16(
          /*neg_a=*/false, a, /*neg_b=*/false, b,
          /*c_mod=*/(short)0, acc, /*reuse_a=*/false, /*reuse_b=*/false);
    }

    // Epilogue: C/D layout -> VGPR r: lanes0-15 M=r, lanes16-31 M=r+8; N=ln.
    if (p < HoWo) {
      float* obase = outBuf + (size_t)n * Cout * HoWo;
#pragma unroll
      for (int r = 0; r < 8; ++r) {
        const int m = m0 + r + lhalf * 8;
        float v = acc[r];
        obase[(size_t)m * HoWo + p] = v > 0.0f ? v : 0.0f;
      }
    }
  }
}

// ------------------------------- launcher ----------------------------------

static inline unsigned cdiv(size_t a, unsigned b) { return (unsigned)((a + b - 1) / b); }

extern "C" void kernel_launch(void* const* d_in, const int* in_sizes, int n_in,
                              void* d_out, int out_size, void* d_ws, size_t ws_size,
                              hipStream_t stream) {
  (void)in_sizes; (void)n_in; (void)out_size; (void)ws_size;
  const float* in_feat = (const float*)d_in[0];   // [2000,1,32,32]
  const float* w1s = (const float*)d_in[1];
  const float* w1n = (const float*)d_in[2];
  const float* w2s = (const float*)d_in[3];
  const float* w2n = (const float*)d_in[4];
  const float* w3s = (const float*)d_in[5];
  const float* w3n = (const float*)d_in[6];
  const int* srcI = (const int*)d_in[7];
  const int* dstI = (const int*)d_in[8];
  float* out = (float*)d_out;                     // [2000,512]

  // workspace layout (floats)
  float* w = (float*)d_ws;
  size_t off = 0;
  float* deg   = w + off; off += NN;                    // [2000]
  float* xcat1 = w + off; off += (size_t)NN * 2048;     // [2000][2][32][32]
  float* xcat2 = w + off; off += (size_t)NN * 14400;    // [2000][64][15][15]
  float* xcat3 = w + off; off += (size_t)NN * 4608;     // [2000][128][6][6]
  float* convb = w + off; off += (size_t)NN * 28800;    // reused: max [2000][32][30][30]
  __bf16* A1 = (__bf16*)(w + off);                      // [32 x 32]
  __bf16* A2 = A1 + 32 * 32;                            // [64 x 576]
  __bf16* A3 = A2 + 64 * 576;                           // [128 x 1152]

  const dim3 B(256);

  // per-call zeroing: degree + aggregation target halves only
  zero_kernel<<<cdiv(NN, 256), B, 0, stream>>>(deg, NN);
  zero_strided_kernel<<<dim3(cdiv(1024, 256), NN), B, 0, stream>>>(xcat1 + 1024, 2048, 1024);
  zero_strided_kernel<<<dim3(cdiv(7200, 256), NN), B, 0, stream>>>(xcat2 + 7200, 14400, 7200);
  zero_strided_kernel<<<dim3(cdiv(2304, 256), NN), B, 0, stream>>>(xcat3 + 2304, 4608, 2304);

  // pack fused [self;neigh] weights to bf16 GEMM-A
  pack_w_kernel<<<dim3(cdiv(32, 256), 32), B, 0, stream>>>(w1s, w1n, A1, 32, 1, 32);
  pack_w_kernel<<<dim3(cdiv(576, 256), 64), B, 0, stream>>>(w2s, w2n, A2, 64, 32, 576);
  pack_w_kernel<<<dim3(cdiv(1152, 256), 128), B, 0, stream>>>(w3s, w3n, A3, 128, 64, 1152);

  degree_kernel<<<cdiv(NE, 256), B, 0, stream>>>(dstI, deg);

  // ---- layer 1: xcat1 = [in_feat ; mean_agg(in_feat)] -> conv -> pool -> xcat2[:32]
  copy_strided_kernel<<<dim3(cdiv(1024, 256), NN), B, 0, stream>>>(in_feat, 1024, xcat1, 2048, 1024);
  edge_agg_kernel<<<dim3(NE, cdiv(1024, 256)), B, 0, stream>>>(in_feat, 1024, 1024,
                                                               xcat1 + 1024, 2048, srcI, dstI);
  scale_agg_kernel<<<dim3(cdiv(1024, 256), NN), B, 0, stream>>>(xcat1 + 1024, 2048, 1024, deg);
  conv_wmma_kernel<2, 32, 32, 32, 32, 30, 30><<<NN, B, 0, stream>>>(xcat1, A1, convb);
  pool_kernel<<<cdiv((size_t)NN * 32 * 15 * 15, 256), B, 0, stream>>>(convb, xcat2, 32, 30, 30,
                                                                      14400, 225);

  // ---- layer 2
  edge_agg_kernel<<<dim3(NE, cdiv(7200, 256)), B, 0, stream>>>(xcat2, 14400, 7200,
                                                               xcat2 + 7200, 14400, srcI, dstI);
  scale_agg_kernel<<<dim3(cdiv(7200, 256), NN), B, 0, stream>>>(xcat2 + 7200, 14400, 7200, deg);
  conv_wmma_kernel<64, 15, 15, 64, 576, 13, 13><<<NN, B, 0, stream>>>(xcat2, A2, convb);
  pool_kernel<<<cdiv((size_t)NN * 64 * 6 * 6, 256), B, 0, stream>>>(convb, xcat3, 64, 13, 13,
                                                                    4608, 36);

  // ---- layer 3 (pool writes straight into d_out: [128,2,2] -> 512 flat)
  edge_agg_kernel<<<dim3(NE, cdiv(2304, 256)), B, 0, stream>>>(xcat3, 4608, 2304,
                                                               xcat3 + 2304, 4608, srcI, dstI);
  scale_agg_kernel<<<dim3(cdiv(2304, 256), NN), B, 0, stream>>>(xcat3 + 2304, 4608, 2304, deg);
  conv_wmma_kernel<128, 6, 6, 128, 1152, 4, 4><<<NN, B, 0, stream>>>(xcat3, A3, convb);
  pool_kernel<<<cdiv((size_t)NN * 128 * 2 * 2, 256), B, 0, stream>>>(convb, out, 128, 4, 4,
                                                                     512, 4);
}
